// DerivNet_9131100472121
// MI455X (gfx1250) — compile-verified, matrix operand order checked by reading, not run on hardware
//
#include <hip/hip_runtime.h>
#include <math.h>

#define NXS 8192
#define DXI 8
#define HID 1024
#define NKT (HID / 32)
// tile config: BM=BN=128, BK=32, 256 threads = 8 waves, wave tile 32x64

typedef __attribute__((ext_vector_type(16))) __bf16 v16bf;
typedef __attribute__((ext_vector_type(8)))  float  v8f;
typedef __attribute__((ext_vector_type(4)))  unsigned int u32x4;
typedef __attribute__((ext_vector_type(8)))  int i32x8;
typedef __attribute__((ext_vector_type(4)))  int i32x4;

#if defined(__HIP_DEVICE_COMPILE__) && \
    __has_builtin(__builtin_amdgcn_tensor_load_to_lds) && \
    __has_builtin(__builtin_amdgcn_s_wait_tensorcnt)
#define USE_TDM 1
#else
#define USE_TDM 0
#endif

struct Frag32 { uint4 lo; uint4 hi; };   // 32 bytes = 16 bf16 per lane

static __device__ __forceinline__ v16bf make_frag(uint4 lo, uint4 hi) {
    Frag32 f{lo, hi};
    return __builtin_bit_cast(v16bf, f);
}

static __device__ __forceinline__ unsigned short f2bf(float f) {
    unsigned int u = __builtin_bit_cast(unsigned int, f);
    u += 0x7FFFu + ((u >> 16) & 1u);          // round-to-nearest-even
    return (unsigned short)(u >> 16);
}
static __device__ __forceinline__ float bf2f(unsigned short h) {
    unsigned int u = ((unsigned int)h) << 16;
    return __builtin_bit_cast(float, u);
}

// full-wave (32-lane) butterfly add via ds_swizzle (group-of-32 xor mode)
static __device__ __forceinline__ float wave_reduce_add(float v) {
    v += __builtin_bit_cast(float, __builtin_amdgcn_ds_swizzle(__builtin_bit_cast(int, v), 0x401F)); // xor 16
    v += __builtin_bit_cast(float, __builtin_amdgcn_ds_swizzle(__builtin_bit_cast(int, v), 0x201F)); // xor 8
    v += __builtin_bit_cast(float, __builtin_amdgcn_ds_swizzle(__builtin_bit_cast(int, v), 0x101F)); // xor 4
    v += __builtin_bit_cast(float, __builtin_amdgcn_ds_swizzle(__builtin_bit_cast(int, v), 0x081F)); // xor 2
    v += __builtin_bit_cast(float, __builtin_amdgcn_ds_swizzle(__builtin_bit_cast(int, v), 0x041F)); // xor 1
    return v;
}

#if USE_TDM
// LDS byte offset of a __shared__ object (addrspace(3) offset)
typedef __attribute__((address_space(3))) const void* as3_cptr;
static __device__ __forceinline__ unsigned lds_off(const void* p) {
    return (unsigned)(unsigned long long)(as3_cptr)p;
}

// Issue one TDM 2-D tile load: 128 rows x 32 bf16, row stride HID elements,
// LDS padding 16B per 64B row (matches As row stride of 5 x uint4 = 80B).
static __device__ __forceinline__ void tdm_load_tile(unsigned long long gaddr,
                                                     unsigned ldsoff) {
    u32x4 g0;
    g0.x = 1u;                                        // count=1 (valid), user mode
    g0.y = ldsoff;                                    // lds_addr (bytes)
    g0.z = (unsigned)gaddr;                           // global_addr[31:0]
    g0.w = (unsigned)(gaddr >> 32) | (2u << 30);      // global_addr[56:32] | type=2

    const unsigned D0 = 1u << 20, D1 = 1u << 20;      // huge tensor dims: no OOB clip
    i32x8 g1;
    g1[0] = (int)((1u << 16)        // data_size = 2 bytes
                | (1u << 20)        // pad_enable
                | (3u << 22)        // pad_interval: 16 DWORDs (64B)
                | (3u << 25));      // pad_amount:   4 DWORDs (16B)
    g1[1] = (int)((D0 & 0xFFFFu) << 16);              // tensor_dim0[15:0]
    g1[2] = (int)((D0 >> 16) | ((D1 & 0xFFFFu) << 16));
    g1[3] = (int)((D1 >> 16) | (32u << 16));          // tile_dim0 = 32 elements
    g1[4] = 128;                                      // tile_dim1 = 128 rows
    g1[5] = (int)HID;                                 // tensor_dim0_stride lo
    g1[6] = 0;
    g1[7] = 0;

    i32x4 z4 = {0, 0, 0, 0};
#if defined(__clang_major__) && __clang_major__ >= 23
    i32x8 z8 = {0, 0, 0, 0, 0, 0, 0, 0};
    __builtin_amdgcn_tensor_load_to_lds(g0, g1, z4, z4, z8, 0);
#else
    __builtin_amdgcn_tensor_load_to_lds(g0, g1, z4, z4, 0);
#endif
}
#endif // USE_TDM

// ---------------------------------------------------------------------------
// Shared WMMA main loop: C[128x128] += A[128xK] * Bt^T, Bt row-major [N][K]
// (rows of length HID). bf16 operands, f32 accum. LDS rows padded to 80B.
// TDM path: double-buffered tensor_load_to_lds, 1 barrier / k-step.
// Fallback: synchronous cooperative copy, 2 barriers / k-step.
// ---------------------------------------------------------------------------
static __device__ __forceinline__ void gemm_tile(
    const unsigned short* __restrict__ A,
    const unsigned short* __restrict__ Bt,
    uint4* As0, uint4* As1, uint4* Bs0, uint4* Bs1,
    v8f acc[2][4])
{
    const int t    = threadIdx.x;
    const int lane = t & 31;
    const int w    = t >> 5;
    const int wm   = w & 3;        // 4 M-groups of 32 rows
    const int wn   = w >> 2;       // 2 N-groups of 64 cols
    const int rsub = lane & 15;
    const int hk   = lane >> 4;    // half-wave selector for K striping

#if USE_TDM
    // prologue: tile 0 into buffer 0
    if (w == 0) {
        tdm_load_tile((unsigned long long)A,  lds_off(As0));
        tdm_load_tile((unsigned long long)Bt, lds_off(Bs0));
        __builtin_amdgcn_s_wait_tensorcnt(0);
    }
    __syncthreads();

    auto step = [&](const uint4* Ac, const uint4* Bc,
                    uint4* An, uint4* Bn, int ktn) {
        if (ktn < NKT && w == 0) {   // stream next K-slab into alternate buffer
            tdm_load_tile((unsigned long long)(A  + ktn * 32), lds_off(An));
            tdm_load_tile((unsigned long long)(Bt + ktn * 32), lds_off(Bn));
        }
        v16bf af[2], bfr[4];
        #pragma unroll
        for (int mi = 0; mi < 2; ++mi) {
            int r = wm * 32 + mi * 16 + rsub;
            af[mi] = make_frag(Ac[r * 5 + hk], Ac[r * 5 + hk + 2]);
        }
        #pragma unroll
        for (int ni = 0; ni < 4; ++ni) {
            int r = wn * 64 + ni * 16 + rsub;
            bfr[ni] = make_frag(Bc[r * 5 + hk], Bc[r * 5 + hk + 2]);
        }
        #pragma unroll
        for (int mi = 0; mi < 2; ++mi)
            #pragma unroll
            for (int ni = 0; ni < 4; ++ni)
                acc[mi][ni] = __builtin_amdgcn_wmma_f32_16x16x32_bf16(
                    false, af[mi], false, bfr[ni], (short)0, acc[mi][ni],
                    false, false);
        if (ktn < NKT) {
            if (w == 0) __builtin_amdgcn_s_wait_tensorcnt(0);
            __syncthreads();
        }
    };

    for (int kt = 0; kt < NKT; kt += 2) {
        step(As0, Bs0, As1, Bs1, kt + 1);   // compute buf0, load buf1
        step(As1, Bs1, As0, Bs0, kt + 2);   // compute buf1, load buf0
    }
#else
    for (int kt = 0; kt < NKT; ++kt) {
        __syncthreads();
        #pragma unroll
        for (int i = 0; i < 2; ++i) {
            int idx = t + i * 256;             // 0..511
            int row = idx >> 2;
            int c   = idx & 3;
            const uint4* ga = (const uint4*)(A  + row * HID + kt * 32) + c;
            const uint4* gb = (const uint4*)(Bt + row * HID + kt * 32) + c;
            As0[row * 5 + c] = *ga;
            Bs0[row * 5 + c] = *gb;
            if (kt + 1 < NKT) {
                __builtin_prefetch((const void*)((const uint4*)(A  + row * HID + (kt + 1) * 32) + c), 0, 1);
                __builtin_prefetch((const void*)((const uint4*)(Bt + row * HID + (kt + 1) * 32) + c), 0, 1);
            }
        }
        __syncthreads();

        v16bf af[2], bfr[4];
        #pragma unroll
        for (int mi = 0; mi < 2; ++mi) {
            int r = wm * 32 + mi * 16 + rsub;
            af[mi] = make_frag(As0[r * 5 + hk], As0[r * 5 + hk + 2]);
        }
        #pragma unroll
        for (int ni = 0; ni < 4; ++ni) {
            int r = wn * 64 + ni * 16 + rsub;
            bfr[ni] = make_frag(Bs0[r * 5 + hk], Bs0[r * 5 + hk + 2]);
        }
        #pragma unroll
        for (int mi = 0; mi < 2; ++mi)
            #pragma unroll
            for (int ni = 0; ni < 4; ++ni)
                acc[mi][ni] = __builtin_amdgcn_wmma_f32_16x16x32_bf16(
                    false, af[mi], false, bfr[ni], (short)0, acc[mi][ni],
                    false, false);
    }
#endif
}

// ---------------------------------------------------------------------------
// prep: W2 -> bf16 (as-is) and bf16 transposed
// ---------------------------------------------------------------------------
__global__ void __launch_bounds__(256) kprep(const float* __restrict__ W2,
                                             unsigned short* __restrict__ w2b,
                                             unsigned short* __restrict__ w2t)
{
    int idx = blockIdx.x * 256 + threadIdx.x;      // 1024*1024 threads
    int r = idx >> 10, c = idx & 1023;
    unsigned short h = f2bf(W2[idx]);
    w2b[idx] = h;
    w2t[c * HID + r] = h;
}

// ---------------------------------------------------------------------------
// k1: z1 = tanh(x @ W1^T + b1); store z1, d1 = 1-z1^2 as bf16
// ---------------------------------------------------------------------------
__global__ void __launch_bounds__(256) k1_input(const float* __restrict__ x,
                                                const float* __restrict__ W1,
                                                const float* __restrict__ b1,
                                                unsigned short* __restrict__ z1b,
                                                unsigned short* __restrict__ d1b)
{
    int idx = blockIdx.x * 256 + threadIdx.x;      // NXS*HID threads
    int n = idx >> 10, j = idx & 1023;
    const float4* xr = (const float4*)(x + n * DXI);
    const float4* wr = (const float4*)(W1 + j * DXI);
    float4 x0 = xr[0], x1 = xr[1], w0 = wr[0], w1 = wr[1];
    float s = x0.x * w0.x + x0.y * w0.y + x0.z * w0.z + x0.w * w0.w
            + x1.x * w1.x + x1.y * w1.y + x1.z * w1.z + x1.w * w1.w + b1[j];
    float z = tanhf(s);
    z1b[idx] = f2bf(z);
    d1b[idx] = f2bf(1.0f - z * z);
}

// ---------------------------------------------------------------------------
// k2: Z2pre = Z1 @ W2^T (+b2) -> z2 = tanh, v = (1-z2^2)*W3[j]
// ---------------------------------------------------------------------------
__global__ void __launch_bounds__(256) k2_hidden(const unsigned short* __restrict__ z1b,
                                                 const unsigned short* __restrict__ w2b,
                                                 const float* __restrict__ b2,
                                                 const float* __restrict__ w3,
                                                 unsigned short* __restrict__ z2b,
                                                 unsigned short* __restrict__ vb)
{
    __shared__ uint4 As[2][128 * 5];
    __shared__ uint4 Bs[2][128 * 5];
    v8f acc[2][4] = {};
    const int n0 = blockIdx.x * 128;
    const int j0 = blockIdx.y * 128;
    gemm_tile(z1b + (size_t)n0 * HID, w2b + (size_t)j0 * HID,
              As[0], As[1], Bs[0], Bs[1], acc);

    const int lane = threadIdx.x & 31;
    const int w    = threadIdx.x >> 5;
    const int wm = w & 3, wn = w >> 2;
    const int rsel = (lane >> 4) * 8;     // C layout: lanes 16-31 hold M+8
    const int cofs = lane & 15;
    #pragma unroll
    for (int mi = 0; mi < 2; ++mi)
        #pragma unroll
        for (int ni = 0; ni < 4; ++ni)
            #pragma unroll
            for (int r = 0; r < 8; ++r) {
                int n = n0 + wm * 32 + mi * 16 + r + rsel;
                int j = j0 + wn * 64 + ni * 16 + cofs;
                float z = tanhf(acc[mi][ni][r] + b2[j]);
                float d = 1.0f - z * z;
                size_t o = (size_t)n * HID + j;
                z2b[o] = f2bf(z);
                vb[o]  = f2bf(d * w3[j]);
            }
}

// ---------------------------------------------------------------------------
// ky: y[n] = z2[n,:] . W3 + b3   (one wave per row)
// ---------------------------------------------------------------------------
__global__ void __launch_bounds__(256) ky_out(const unsigned short* __restrict__ z2b,
                                              const float* __restrict__ w3,
                                              const float* __restrict__ b3,
                                              float* __restrict__ y)
{
    int wave = threadIdx.x >> 5, lane = threadIdx.x & 31;
    int n = blockIdx.x * 8 + wave;
    float s = 0.0f;
    #pragma unroll 4
    for (int i = 0; i < HID / 32; ++i) {
        int j = lane + i * 32;
        s += bf2f(z2b[(size_t)n * HID + j]) * w3[j];
    }
    s = wave_reduce_add(s);
    if (lane == 0) y[n] = s + b3[0];
}

// ---------------------------------------------------------------------------
// k3: A = V @ W2 (via W2^T tiles); G = A .* d1  -> bf16
// ---------------------------------------------------------------------------
__global__ void __launch_bounds__(256) k3_back(const unsigned short* __restrict__ vb,
                                               const unsigned short* __restrict__ w2t,
                                               const unsigned short* __restrict__ d1b,
                                               unsigned short* __restrict__ gb)
{
    __shared__ uint4 As[2][128 * 5];
    __shared__ uint4 Bs[2][128 * 5];
    v8f acc[2][4] = {};
    const int n0 = blockIdx.x * 128;
    const int j0 = blockIdx.y * 128;
    gemm_tile(vb + (size_t)n0 * HID, w2t + (size_t)j0 * HID,
              As[0], As[1], Bs[0], Bs[1], acc);

    const int lane = threadIdx.x & 31;
    const int w    = threadIdx.x >> 5;
    const int wm = w & 3, wn = w >> 2;
    const int rsel = (lane >> 4) * 8;
    const int cofs = lane & 15;
    #pragma unroll
    for (int mi = 0; mi < 2; ++mi)
        #pragma unroll
        for (int ni = 0; ni < 4; ++ni)
            #pragma unroll
            for (int r = 0; r < 8; ++r) {
                int n = n0 + wm * 32 + mi * 16 + r + rsel;
                int j = j0 + wn * 64 + ni * 16 + cofs;
                size_t o = (size_t)n * HID + j;
                float g = acc[mi][ni][r] * bf2f(d1b[o]);
                gb[o] = f2bf(g);
            }
}

// ---------------------------------------------------------------------------
// k4: dydx[k][n] = sum_j G[n,j] * W1[j,k]   (one wave per row, 8 outputs)
// ---------------------------------------------------------------------------
__global__ void __launch_bounds__(256) k4_dydx(const unsigned short* __restrict__ gb,
                                               const float* __restrict__ W1,
                                               float* __restrict__ dydx)
{
    int wave = threadIdx.x >> 5, lane = threadIdx.x & 31;
    int n = blockIdx.x * 8 + wave;
    float acc[8] = {};
    for (int i = 0; i < HID / 32; ++i) {
        int j = lane + i * 32;
        float g = bf2f(gb[(size_t)n * HID + j]);
        const float4* wr = (const float4*)(W1 + j * DXI);
        float4 a = wr[0], b = wr[1];
        acc[0] += g * a.x; acc[1] += g * a.y; acc[2] += g * a.z; acc[3] += g * a.w;
        acc[4] += g * b.x; acc[5] += g * b.y; acc[6] += g * b.z; acc[7] += g * b.w;
    }
    #pragma unroll
    for (int k = 0; k < DXI; ++k) {
        float s = wave_reduce_add(acc[k]);
        if (lane == 0) dydx[(size_t)k * NXS + n] = s;
    }
}

// ---------------------------------------------------------------------------
extern "C" void kernel_launch(void* const* d_in, const int* in_sizes, int n_in,
                              void* d_out, int out_size, void* d_ws, size_t ws_size,
                              hipStream_t stream)
{
    const float* x  = (const float*)d_in[0];
    const float* W1 = (const float*)d_in[1];
    const float* b1 = (const float*)d_in[2];
    const float* W2 = (const float*)d_in[3];
    const float* b2 = (const float*)d_in[4];
    const float* W3 = (const float*)d_in[5];
    const float* b3 = (const float*)d_in[6];
    float* out = (float*)d_out;               // [0,8192): y ; [8192,73728): dydx

    // workspace layout (bytes), all bf16 buffers
    char* ws = (char*)d_ws;
    const size_t szNH = (size_t)NXS * HID * 2;   // 16 MB
    const size_t szHH = (size_t)HID * HID * 2;   //  2 MB
    unsigned short* z1b = (unsigned short*)(ws);
    unsigned short* w2b = (unsigned short*)(ws + szNH);
    unsigned short* w2t = (unsigned short*)(ws + szNH + szHH);
    unsigned short* d1b = (unsigned short*)(ws + szNH + 2 * szHH);
    unsigned short* z2b = (unsigned short*)(ws + 2 * szNH + 2 * szHH);
    unsigned short* vb  = (unsigned short*)(ws + 3 * szNH + 2 * szHH);
    unsigned short* gb  = z1b;   // alias: z1 is dead after k2 completes (stream order)

    kprep<<<(HID * HID) / 256, 256, 0, stream>>>(W2, w2b, w2t);
    k1_input<<<(NXS * HID) / 256, 256, 0, stream>>>(x, W1, b1, z1b, d1b);

    dim3 gg(NXS / 128, HID / 128);               // 64 x 8 workgroups
    k2_hidden<<<gg, 256, 0, stream>>>(z1b, w2b, b2, W3, z2b, vb);
    ky_out<<<NXS / 8, 256, 0, stream>>>(z2b, W3, b3, out);
    k3_back<<<gg, 256, 0, stream>>>(vb, w2t, d1b, gb);
    k4_dydx<<<NXS / 8, 256, 0, stream>>>(gb, W1, out + NXS);
}
